// Qwen3VLMoeTextExperts_86792699117782
// MI455X (gfx1250) — matrix operand order repaired and checked by reading, not compiled, not run on hardware
//
#include <hip/hip_runtime.h>

// Qwen3VLMoeTextExperts — dense MoE experts (E=8, H=2048, I=768, T=4096)
// Pass 0: Xb = bf16(hidden_states)                       (ws + 48MB)
// Pass 1: interw[e,t,i] = rw[t,e] * up * silu(gate)      (bf16, ws)
// Pass 2: out[t,h]      = sum_e,i interw[e,t,i]*down[e,i,h]  (f32)
// WMMA bf16 16x16x32, f32 accumulate; LDS staged via
// global_load_async_to_lds_b128 (ASYNCcnt) where the source is bf16.

#define E_EXP   8
#define H_DIM   2048
#define I_DIM   768
#define I2_DIM  1536
#define T_TOK   4096
#define TM      128
#define TN      64
#define TK      32
#define LDPAD   40          // ushort stride (80B row): 16B-aligned b128 targets

typedef __attribute__((ext_vector_type(16))) __bf16 v16bf;
typedef __attribute__((ext_vector_type(8)))  float  v8f;

union FragU {
    unsigned int u[8];
    v16bf        v;
};

__device__ __forceinline__ unsigned short f2bf(float f) {
    unsigned int u = __float_as_uint(f);
    u += 0x7FFFu + ((u >> 16) & 1u);      // round-to-nearest-even
    return (unsigned short)(u >> 16);
}

// silu(g)*u with fast hardware reciprocal (v_rcp_f32), no IEEE div expansion
__device__ __forceinline__ float silu_mul(float g, float u) {
    float sig = __builtin_amdgcn_rcpf(1.0f + __expf(-g));
    return u * g * sig;
}

__device__ __forceinline__ v8f wmma_bf16(const FragU& a, const FragU& b, v8f c) {
    return __builtin_amdgcn_wmma_f32_16x16x32_bf16(
        false, a.v, false, b.v, (short)0, c, false, false);
}

// async 16B copy: global (bf16 data) -> LDS, tracked by ASYNCcnt
__device__ __forceinline__ void async_b128(const unsigned short* lds_dst,
                                           const unsigned short* gsrc) {
    unsigned loff = (unsigned)(uintptr_t)lds_dst;   // low 32b of LDS aperture = offset
    asm volatile("global_load_async_to_lds_b128 %0, %1, off"
                 :: "v"(loff), "v"(gsrc) : "memory");
}
__device__ __forceinline__ void wait_async0() {
    asm volatile("s_wait_asynccnt 0x0" ::: "memory");
}

__device__ __forceinline__ FragU load_a_frag(const unsigned short* s, int m, int half) {
    FragU a;
    const int kb = 8 * half;
    #pragma unroll
    for (int i = 0; i < 4; ++i) {
        a.u[i]     = *(const unsigned int*)&s[m * LDPAD + kb + 2 * i];
        a.u[i + 4] = *(const unsigned int*)&s[m * LDPAD + 16 + kb + 2 * i];
    }
    return a;
}
__device__ __forceinline__ FragU load_b_frag(const unsigned short* s, int n, int half) {
    FragU b;
    const int kb = 8 * half;
    #pragma unroll
    for (int i = 0; i < 4; ++i) {
        b.u[i]     = *(const unsigned int*)&s[n * LDPAD + kb + 2 * i];
        b.u[i + 4] = *(const unsigned int*)&s[n * LDPAD + 16 + kb + 2 * i];
    }
    return b;
}

// ---------------------------------------------------------------------------
// Kernel 0: f32 -> bf16 pre-convert of hidden_states
// ---------------------------------------------------------------------------
__global__ __launch_bounds__(256) void k_cvt(const float* __restrict__ in,
                                             unsigned short* __restrict__ outp) {
    int i = blockIdx.x * 256 + threadIdx.x;           // pair index
    float2 f = ((const float2*)in)[i];
    ((unsigned int*)outp)[i] =
        (unsigned int)f2bf(f.x) | ((unsigned int)f2bf(f.y) << 16);
}

// ---------------------------------------------------------------------------
// Kernel 1: gate/up GEMM + SiLU + routing-weight fusion -> bf16 workspace
// grid: (I/64, T/128, E); block 256 = 8 wave32; wave tile = 32x32 (x2 gate/up)
// ---------------------------------------------------------------------------
__global__ __launch_bounds__(256) void k_gateup(
    const unsigned short* __restrict__ Xb,   // [T, H] bf16
    const float* __restrict__ RW,            // [T, E]
    const float* __restrict__ GUP,           // [E, H, 2I]
    unsigned short* __restrict__ interw)     // [E, T, I] bf16
{
    __shared__ unsigned short sA [TM * LDPAD];
    __shared__ unsigned short sBg[TN * LDPAD];
    __shared__ unsigned short sBu[TN * LDPAD];

    const int d0  = blockIdx.x * TN;
    const int t0  = blockIdx.y * TM;
    const int e   = blockIdx.z;
    const int tid  = threadIdx.x;
    const int lane = tid & 31;
    const int wave = tid >> 5;
    const int wmB  = (wave & 3) * 32;        // wave m-offset (2 subtiles)
    const int wn   = (wave >> 2) * 32;       // wave n-offset (2 subtiles)
    const int half = lane >> 4;
    const int l16  = lane & 15;

    v8f g00 = {}, g01 = {}, g10 = {}, g11 = {};
    v8f u00 = {}, u01 = {}, u10 = {}, u11 = {};

    const float* Wg = GUP + (size_t)e * H_DIM * I2_DIM;

    const int kq   = (tid & 3) * 8;          // 8 bf16 = 16B per async lane
    const int rowA = tid >> 2;               // 0..63

    for (int hk = 0; hk < H_DIM; hk += TK) {
        // ---- A tile: async bf16 global -> LDS (128x32, 2 passes) ----
        #pragma unroll
        for (int p = 0; p < 2; ++p) {
            const int r = rowA + p * 64;
            async_b128(&sA[r * LDPAD + kq],
                       Xb + (size_t)(t0 + r) * H_DIM + hk + kq);
        }
        // ---- B gate/up tiles: f32 -> bf16, column-major in LDS ----
        {
            const int n2 = (tid & 31) * 2;
            int k = tid >> 5;
            #pragma unroll
            for (int p = 0; p < 4; ++p, k += 8) {
                const float* base = Wg + (size_t)(hk + k) * I2_DIM + d0 + n2;
                float2 g = *(const float2*)base;
                sBg[ n2      * LDPAD + k] = f2bf(g.x);
                sBg[(n2 + 1) * LDPAD + k] = f2bf(g.y);
                float2 u = *(const float2*)(base + I_DIM);
                sBu[ n2      * LDPAD + k] = f2bf(u.x);
                sBu[(n2 + 1) * LDPAD + k] = f2bf(u.y);
            }
        }
        wait_async0();
        __syncthreads();

        FragU a0 = load_a_frag(sA, wmB + l16, half);
        FragU a1 = load_a_frag(sA, wmB + 16 + l16, half);

        FragU b = load_b_frag(sBg, wn + l16, half);
        g00 = wmma_bf16(a0, b, g00);  g10 = wmma_bf16(a1, b, g10);
        b = load_b_frag(sBg, wn + 16 + l16, half);
        g01 = wmma_bf16(a0, b, g01);  g11 = wmma_bf16(a1, b, g11);
        b = load_b_frag(sBu, wn + l16, half);
        u00 = wmma_bf16(a0, b, u00);  u10 = wmma_bf16(a1, b, u10);
        b = load_b_frag(sBu, wn + 16 + l16, half);
        u01 = wmma_bf16(a0, b, u01);  u11 = wmma_bf16(a1, b, u11);
        __syncthreads();
    }

    // ---- epilogue: rw * up * silu(gate) -> bf16 workspace ----
    const size_t baseOut = (size_t)e * T_TOK * I_DIM;
    #pragma unroll
    for (int r = 0; r < 8; ++r) {
        const int ta = t0 + wmB + r + 8 * half;        // m-subtile 0
        const int tb = ta + 16;                        // m-subtile 1
        const float wa = RW[(size_t)ta * E_EXP + e];
        const float wb = RW[(size_t)tb * E_EXP + e];
        const size_t ca = baseOut + (size_t)ta * I_DIM + d0 + wn + l16;
        const size_t cb = baseOut + (size_t)tb * I_DIM + d0 + wn + l16;

        interw[ca]      = f2bf(silu_mul(g00[r], u00[r]) * wa);
        interw[ca + 16] = f2bf(silu_mul(g01[r], u01[r]) * wa);
        interw[cb]      = f2bf(silu_mul(g10[r], u10[r]) * wb);
        interw[cb + 16] = f2bf(silu_mul(g11[r], u11[r]) * wb);
    }
}

// ---------------------------------------------------------------------------
// Kernel 2: down GEMM, expert-sum in registers -> f32 output
// grid: (H/64, T/128); block 256 = 8 wave32; wave tile 32x32
// ---------------------------------------------------------------------------
__global__ __launch_bounds__(256) void k_down(
    const unsigned short* __restrict__ interw, // [E, T, I] bf16
    const float* __restrict__ DWN,             // [E, I, H]
    float* __restrict__ out)                   // [T, H]
{
    __shared__ unsigned short sA[TM * LDPAD];
    __shared__ unsigned short sB[TN * LDPAD];

    const int h0  = blockIdx.x * TN;
    const int t0  = blockIdx.y * TM;
    const int tid  = threadIdx.x;
    const int lane = tid & 31;
    const int wave = tid >> 5;
    const int wmB  = (wave & 3) * 32;
    const int wn   = (wave >> 2) * 32;
    const int half = lane >> 4;
    const int l16  = lane & 15;

    v8f c00 = {}, c01 = {}, c10 = {}, c11 = {};

    const int kq   = (tid & 3) * 8;
    const int rowA = tid >> 2;

    for (int e = 0; e < E_EXP; ++e) {
        const unsigned short* Ain = interw + (size_t)e * T_TOK * I_DIM;
        const float*          Bdn = DWN    + (size_t)e * I_DIM * H_DIM;

        for (int kk = 0; kk < I_DIM; kk += TK) {
            // ---- A tile: async bf16 workspace -> LDS ----
            #pragma unroll
            for (int p = 0; p < 2; ++p) {
                const int r = rowA + p * 64;
                async_b128(&sA[r * LDPAD + kq],
                           Ain + (size_t)(t0 + r) * I_DIM + kk + kq);
            }
            // ---- B tile: f32 -> bf16, column-major ----
            {
                const int n2 = (tid & 31) * 2;
                int k = tid >> 5;
                #pragma unroll
                for (int p = 0; p < 4; ++p, k += 8) {
                    float2 s = *(const float2*)
                        (Bdn + (size_t)(kk + k) * H_DIM + h0 + n2);
                    sB[ n2      * LDPAD + k] = f2bf(s.x);
                    sB[(n2 + 1) * LDPAD + k] = f2bf(s.y);
                }
            }
            wait_async0();
            __syncthreads();

            FragU a0 = load_a_frag(sA, wmB + l16, half);
            FragU a1 = load_a_frag(sA, wmB + 16 + l16, half);
            FragU b  = load_b_frag(sB, wn + l16, half);
            c00 = wmma_bf16(a0, b, c00);  c10 = wmma_bf16(a1, b, c10);
            b = load_b_frag(sB, wn + 16 + l16, half);
            c01 = wmma_bf16(a0, b, c01);  c11 = wmma_bf16(a1, b, c11);
            __syncthreads();
        }
    }

    #pragma unroll
    for (int r = 0; r < 8; ++r) {
        const int ta = t0 + wmB + r + 8 * half;
        const int tb = ta + 16;
        float* pa = out + (size_t)ta * H_DIM + h0 + wn + l16;
        float* pb = out + (size_t)tb * H_DIM + h0 + wn + l16;
        pa[0]  = c00[r];
        pa[16] = c01[r];
        pb[0]  = c10[r];
        pb[16] = c11[r];
    }
}

// ---------------------------------------------------------------------------
extern "C" void kernel_launch(void* const* d_in, const int* in_sizes, int n_in,
                              void* d_out, int out_size, void* d_ws, size_t ws_size,
                              hipStream_t stream) {
    const float* X   = (const float*)d_in[0];   // hidden_states  [B,S,H]
    const float* RW  = (const float*)d_in[1];   // routing_weights [T,E]
    // d_in[2] = router_indices: unused (reference is dense over all experts)
    const float* GUP = (const float*)d_in[3];   // gate_up_proj [E,H,2I]
    const float* DWN = (const float*)d_in[4];   // down_proj    [E,I,H]
    float* out = (float*)d_out;

    unsigned short* interw = (unsigned short*)d_ws;                    // 48 MB bf16
    unsigned short* Xb     = (unsigned short*)((char*)d_ws + (size_t)48 * 1024 * 1024);

    k_cvt<<<T_TOK * H_DIM / 512, 256, 0, stream>>>(X, Xb);             // 16384 blocks

    dim3 g1(I_DIM / TN, T_TOK / TM, E_EXP);     // (12, 32, 8)
    k_gateup<<<g1, 256, 0, stream>>>(Xb, RW, GUP, interw);

    dim3 g2(H_DIM / TN, T_TOK / TM);            // (32, 32)
    k_down<<<g2, 256, 0, stream>>>(interw, DWN, out);
}
